// PetForwardRadon_51049981280696
// MI455X (gfx1250) — compile-verified, matrix operand order checked by reading, not run on hardware
//
#include <hip/hip_runtime.h>
#include <hip/hip_bf16.h>

typedef __attribute__((ext_vector_type(16))) _Float16     v16h;
typedef __attribute__((ext_vector_type(8)))  float        v8f;
typedef __attribute__((ext_vector_type(4)))  unsigned int v4u;
typedef __attribute__((ext_vector_type(8)))  int          v8i;
typedef __attribute__((ext_vector_type(4)))  int          v4i;

#define NP        300            // padded image / det_count / n_angles / n_t
#define IMGN      256
#define PADW      22
#define CTR       149.5f
#define SIG_PSF   0.60056120f    // (4/(4*sqrt(ln2)))/2
#define SIG_ATT   1.20112240f    // (4/(2*sqrt(ln2)))/2
#define SIG_MM    1.20112240f    // fwhm/(4*sqrt(ln2))
#define THSTEP    (3.14159265358979f / 299.0f)
#define D2R       (3.14159265358979f / 180.0f)
#define ANG_PER_WG 2
#define LDS_BYTES  (NP * NP * 2) // 180000 B fp16 image, fits 320KB WGP LDS

// ---------------- stage 1a: horizontal blur ----------------
__global__ void pet_blurH(const float* __restrict__ img,
                          const float* __restrict__ att,
                          float* __restrict__ w1) {
    int id = blockIdx.x * blockDim.x + threadIdx.x;
    if (id >= 16 * NP * NP) return;
    int x = id % NP, y = (id / NP) % NP, im = id / (NP * NP);
    int b = im & 7;
    if (im < 8) {
        if (x >= IMGN || y >= IMGN) return;
        float acc = 0.f, wsum = 0.f;
        const float inv2s2 = 1.0f / (2.0f * SIG_PSF * SIG_PSF);
#pragma unroll
        for (int i = -3; i <= 3; ++i) {
            float w = __expf(-(float)(i * i) * inv2s2);
            int j = x + i; j = j < 0 ? -j : (j > IMGN - 1 ? 2 * IMGN - 2 - j : j);
            acc += w * img[(b * IMGN + y) * IMGN + j]; wsum += w;
        }
        w1[id] = acc / wsum;
    } else {
        float acc = 0.f, wsum = 0.f;
        const float inv2s2 = 1.0f / (2.0f * SIG_ATT * SIG_ATT);
#pragma unroll
        for (int i = -5; i <= 5; ++i) {
            float w = __expf(-(float)(i * i) * inv2s2);
            int j = x + i; j = j < 0 ? -j : (j > NP - 1 ? 2 * NP - 2 - j : j);
            float v = 0.f;
            if (j >= PADW && j < PADW + IMGN && y >= PADW && y < PADW + IMGN)
                v = att[(b * IMGN + (y - PADW)) * IMGN + (j - PADW)];
            acc += w * v; wsum += w;
        }
        w1[id] = acc / wsum;
    }
}

// ------- stage 1b: vertical blur + pad + circle mask + fp16 pack -------
__global__ void pet_blurV(const float* __restrict__ w1,
                          _Float16* __restrict__ w2) {
    int id = blockIdx.x * blockDim.x + threadIdx.x;
    if (id >= 16 * NP * NP) return;
    int x = id % NP, y = (id / NP) % NP, im = id / (NP * NP);
    float cx = (float)x - CTR, cy = (float)y - CTR;
    float mask = (cx * cx + cy * cy <= 22500.0f) ? 1.0f : 0.0f;
    float val = 0.f;
    if (im < 8) {
        if (x >= PADW && x < PADW + IMGN && y >= PADW && y < PADW + IMGN) {
            int sx = x - PADW, sy = y - PADW;
            float acc = 0.f, wsum = 0.f;
            const float inv2s2 = 1.0f / (2.0f * SIG_PSF * SIG_PSF);
#pragma unroll
            for (int i = -3; i <= 3; ++i) {
                float w = __expf(-(float)(i * i) * inv2s2);
                int j = sy + i; j = j < 0 ? -j : (j > IMGN - 1 ? 2 * IMGN - 2 - j : j);
                acc += w * w1[(im * NP + j) * NP + sx]; wsum += w;
            }
            val = acc / wsum;
        }
    } else {
        float acc = 0.f, wsum = 0.f;
        const float inv2s2 = 1.0f / (2.0f * SIG_ATT * SIG_ATT);
#pragma unroll
        for (int i = -5; i <= 5; ++i) {
            float w = __expf(-(float)(i * i) * inv2s2);
            int j = y + i; j = j < 0 ? -j : (j > NP - 1 ? 2 * NP - 2 - j : j);
            acc += w * w1[(im * NP + j) * NP + x]; wsum += w;
        }
        val = (acc / wsum) * 0.01f;   // radon_sinogram(att * 0.01)
    }
    w2[id] = (_Float16)(val * mask);
}

// ---------------- stage 2: WMMA radon + attenuation fusion ----------------
__device__ __forceinline__ float pet_fetch(const _Float16* sh, int y, int x) {
    int yc = min(max(y, 0), NP - 1), xc = min(max(x, 0), NP - 1);
    float v = (float)sh[yc * NP + xc];
    return ((unsigned)x < (unsigned)NP && (unsigned)y < (unsigned)NP) ? v : 0.0f;
}

__device__ __forceinline__ float pet_bilin(const _Float16* sh, float x, float y) {
    float xf = floorf(x), yf = floorf(y);
    int xi = (int)xf, yi = (int)yf;
    float wx = x - xf, wy = y - yf;
    float v00 = pet_fetch(sh, yi, xi),     v01 = pet_fetch(sh, yi, xi + 1);
    float v10 = pet_fetch(sh, yi + 1, xi), v11 = pet_fetch(sh, yi + 1, xi + 1);
    return (v00 * (1.f - wx) + v01 * wx) * (1.f - wy)
         + (v10 * (1.f - wx) + v11 * wx) * wy;
}

// TDM: DMA one 300x300 fp16 image (tile == tensor) from global into LDS base 0.
// D# bit layout per CDNA5 ISA ch.8 (group0: count/lds_addr/global_addr/type,
// group1: data_size + dims + strides).  Issued by one wave; EXEC is ignored.
__device__ __forceinline__ void pet_tdm_load(const _Float16* src) {
    unsigned long long ga = (unsigned long long)(size_t)src;
    v4u g0;
    g0[0] = 1u;                                        // count=1 (valid user D#)
    g0[1] = 0u;                                        // lds_addr = 0
    g0[2] = (unsigned)(ga & 0xFFFFFFFFu);              // global_addr[31:0]
    g0[3] = (unsigned)((ga >> 32) & 0x01FFFFFFu)       // global_addr[56:32]
          | (2u << 30);                                // type = 2 ("image")
    v8i g1;
    g1[0] = (int)(1u << 16);                           // wg_mask=0, data_size=1 (2B)
    g1[1] = (int)((NP & 0xFFFFu) << 16);               // tensor_dim0[15:0] << 16
    g1[2] = (int)((NP >> 16) | ((NP & 0xFFFFu) << 16));// dim0 hi | dim1 lo
    g1[3] = (int)((NP >> 16) | ((unsigned)NP << 16));  // dim1 hi | tile_dim0=300
    g1[4] = (int)NP;                                   // tile_dim1=300, tile_dim2=0
    g1[5] = (int)NP;                                   // tensor_dim0_stride = 300
    g1[6] = 0;                                         // stride0 hi | stride1 lo
    g1[7] = 0;                                         // stride1 hi
    v4i z4 = {0, 0, 0, 0};
#if __has_include(<hip/amd_detail/amd_gfx1250_TDM.h>)
    v8i z8 = {0, 0, 0, 0, 0, 0, 0, 0};
    __builtin_amdgcn_tensor_load_to_lds(g0, g1, z4, z4, z8, 0);
#else
    __builtin_amdgcn_tensor_load_to_lds(g0, g1, z4, z4, 0);
#endif
}

// Pack 16 fp16 ray samples into the 16x32 f16 A layout for one K-block.
__device__ __forceinline__ v16h pet_build_a(const _Float16* sh, int tb, int koff,
                                            float xs, float ys, float sa, float ca) {
    v16h av;
#pragma unroll
    for (int e = 0; e < 16; ++e) {
        int kk = e + (e & 8) + koff;                   // ISA A-layout K index
        float tm = (tb + kk < NP) ? 1.0f : 0.0f;       // t >= 300 contributes 0
        float tv = (float)(tb + kk) - CTR;
        av[e] = (_Float16)(pet_bilin(sh, xs - tv * sa, ys + tv * ca) * tm);
    }
    return av;
}

// Line integral over t for 16 detectors. Single accumulator chain: the 16
// bilinear gathers between consecutive WMMAs already cover the WMMA->WMMA
// hazard window, and one chain keeps VGPR pressure low (no scratch spills).
__device__ __forceinline__ v8f pet_project(const _Float16* sh, int chunk, int mrow,
                                           int koff, float ca, float sa,
                                           v16h ones) {
    v8f acc = {0.f, 0.f, 0.f, 0.f, 0.f, 0.f, 0.f, 0.f};
    float s  = (float)(chunk * 16 + mrow) - CTR;
    float xs = s * ca + CTR;
    float ys = s * sa + CTR;
#pragma unroll 1
    for (int tb = 0; tb < 320; tb += 32) {
        v16h av = pet_build_a(sh, tb, koff, xs, ys, sa, ca);
        acc = __builtin_amdgcn_wmma_f32_16x16x32_f16(
            false, av, false, ones, (short)0, acc, false, false);
    }
    return acc;
}

// One WG: 8 waves, one batch, ANG_PER_WG angles. Wave w owns detector
// chunks {w, w+8, w+16} (19 chunks of 16 cover 300 detectors).
// Phase-1 emission sums round-trip through S (L2-resident) instead of being
// held in VGPRs across phase 2 -> no register spills in the gather loop.
__global__ void pet_radon_wmma(const _Float16* __restrict__ w2,
                               float* __restrict__ S) {
    extern __shared__ char smem[];
    const _Float16* sh = (const _Float16*)smem;
    int tid  = threadIdx.x;
    int lane = tid & 31, wv = tid >> 5;
    int mrow = lane & 15;
    int koff = (lane >> 4) * 8;          // lanes 16..31 start at K=8 / K=24
    int b  = blockIdx.x / (NP / ANG_PER_WG);
    int ag = blockIdx.x % (NP / ANG_PER_WG);

    v16h ones;
#pragma unroll
    for (int e = 0; e < 16; ++e) ones[e] = (_Float16)1.0f;

    const _Float16* emisImg = w2 + (size_t)b * NP * NP;
    const _Float16* attImg  = w2 + (size_t)(8 + b) * NP * NP;

    // warm the attenuation image into L2 while phase 1 runs
    for (int i = tid; i < (NP * NP) / 64; i += blockDim.x)
        __builtin_prefetch((const char*)attImg + i * 128, 0, 0);

    if (wv == 0) {                       // one wave drives the TDM
        pet_tdm_load(emisImg);
        __builtin_amdgcn_s_wait_tensorcnt(0);
    }
    __syncthreads();

    // ---------- phase 1: emission projections -> S ----------
#pragma unroll 1
    for (int ai = 0; ai < ANG_PER_WG; ++ai) {
        int a = ag * ANG_PER_WG + ai;
        float ca, sa;
        __sincosf((float)a * THSTEP, &sa, &ca);
#pragma unroll 1
        for (int j = 0; j < 3; ++j) {
            int chunk = wv + 8 * j;                 // wave-uniform guard
            if (chunk < 19) {
                v8f acc = pet_project(sh, chunk, mrow, koff, ca, sa, ones);
                if (mrow == 0) {   // lanes 0 / 16 hold column N=0: M=0..7 / 8..15
#pragma unroll
                    for (int i = 0; i < 8; ++i) {
                        int det = chunk * 16 + koff + i;
                        if (det < NP)
                            S[((size_t)b * NP + a) * NP + det] = acc[i];
                    }
                }
            }
        }
    }

    __syncthreads();
    if (wv == 0) {
        pet_tdm_load(attImg);
        __builtin_amdgcn_s_wait_tensorcnt(0);
    }
    __syncthreads();

    // ---------- phase 2: attenuation projections + fuse ----------
#pragma unroll 1
    for (int ai = 0; ai < ANG_PER_WG; ++ai) {
        int a = ag * ANG_PER_WG + ai;
        float ca, sa;
        __sincosf((float)a * THSTEP, &sa, &ca);
#pragma unroll 1
        for (int j = 0; j < 3; ++j) {
            int chunk = wv + 8 * j;
            if (chunk < 19) {
                v8f acc = pet_project(sh, chunk, mrow, koff, ca, sa, ones);
                if (mrow == 0) {
#pragma unroll
                    for (int i = 0; i < 8; ++i) {
                        int det = chunk * 16 + koff + i;
                        if (det < NP) {
                            size_t idx = ((size_t)b * NP + a) * NP + det;
                            float e = S[idx];       // same-wave store->load: ordered
                            S[idx] = e * __expf(-acc[i] * 2.0f);
                        }
                    }
                }
            }
        }
    }
}

// -------- stage 3: per-angle detector-axis 7-tap blur + scale --------
__global__ void pet_colblur(const float* __restrict__ S,
                            const float* __restrict__ scale,
                            float* __restrict__ out) {
    int id = blockIdx.x * blockDim.x + threadIdx.x;
    if (id >= 8 * NP * NP) return;
    int a = id % NP, d = (id / NP) % NP, b = id / (NP * NP);
    float th = (float)a * THSTEP;
    float bw = 2.0f * (fabsf(__cosf(th * D2R)) + fabsf(__sinf(th * D2R)));
    float sig = SIG_MM / bw;
    float inv2s2 = 1.0f / (2.0f * sig * sig);
    float acc = 0.f, wsum = 0.f;
#pragma unroll
    for (int i = -3; i <= 3; ++i) {
        float w = __expf(-(float)(i * i) * inv2s2);
        int j = d + i; j = j < 0 ? -j : (j > NP - 1 ? 2 * NP - 2 - j : j);
        acc += w * S[((size_t)b * NP + a) * NP + j]; wsum += w;
    }
    out[((size_t)b * NP + d) * NP + a] = (acc / wsum) * scale[b];
}

extern "C" void kernel_launch(void* const* d_in, const int* in_sizes, int n_in,
                              void* d_out, int out_size, void* d_ws, size_t ws_size,
                              hipStream_t stream) {
    const float* image  = (const float*)d_in[0];   // [8,1,256,256]
    const float* attmap = (const float*)d_in[1];   // [8,1,256,256]
    const float* scale  = (const float*)d_in[2];   // [8]
    float* out = (float*)d_out;                    // [8,1,300,300]

    char* ws = (char*)d_ws;
    _Float16* W2 = (_Float16*)ws;                       // 16 fp16 images: 2,880,000 B
    float*    W1 = (float*)(ws + 2880000);              // blur scratch:   5,760,000 B
    float*    S  = (float*)(ws + 2880000 + 5760000);    // pre-blur sino:  2,880,000 B

    int n1 = 16 * NP * NP;
    pet_blurH<<<(n1 + 255) / 256, 256, 0, stream>>>(image, attmap, W1);
    pet_blurV<<<(n1 + 255) / 256, 256, 0, stream>>>(W1, W2);

    (void)hipFuncSetAttribute((const void*)pet_radon_wmma,
                              hipFuncAttributeMaxDynamicSharedMemorySize, LDS_BYTES);
    pet_radon_wmma<<<8 * (NP / ANG_PER_WG), 256, LDS_BYTES, stream>>>(W2, S);

    int n3 = 8 * NP * NP;
    pet_colblur<<<(n3 + 255) / 256, 256, 0, stream>>>(S, scale, out);
}
